// SelfAttention2D_76227079569760
// MI455X (gfx1250) — matrix operand order repaired and checked by reading, not compiled, not run on hardware
//
#include <hip/hip_runtime.h>
#include <hip/hip_bf16.h>

typedef __attribute__((ext_vector_type(16))) __bf16 v16bf;
typedef __attribute__((ext_vector_type(8)))  float  v8f;

static __device__ __forceinline__ unsigned short f32_to_bf16_bits(float f) {
  __bf16 h = (__bf16)f;
  return __builtin_bit_cast(unsigned short, h);
}

// Pack 16 u16 (register array, fully unrolled) into two 16B stores.
static __device__ __forceinline__ void store_frag(unsigned short* dst,
                                                  const unsigned short* v) {
  union { unsigned int u[8]; uint4 q[2]; } pk;
#pragma unroll
  for (int j = 0; j < 8; ++j)
    pk.u[j] = (unsigned int)v[2 * j] | ((unsigned int)v[2 * j + 1] << 16);
  ((uint4*)dst)[0] = pk.q[0];
  ((uint4*)dst)[1] = pk.q[1];
}

// -------------------------------------------------------------------------
// Projection with fragment-swizzled output. One wave = 32(o) x 32(n) tile,
// 4 WMMA accumulators, K-loop over C=512. Result goes through a per-wave
// LDS tile, then out as WMMA-fragment tiles: [tile][lane(32)][i(16)] bf16,
// i.e. 32 contiguous bytes per lane -> the attention kernel loads every
// fragment with two b128s.
// mode 0: Q  A-frags  [B][ntile 256][kb 2]   (scale=1/8 folded)
// mode 1: K  B-frags  [B][mtile 256][kb 2]   (k=o, n=m)
// mode 2: Vt B-frags  [B][km 128][cn 32]     (k=m, n=c)
// -------------------------------------------------------------------------
__global__ __launch_bounds__(256) void proj_kernel(
    const float* __restrict__ X, const float* __restrict__ W,
    const float* __restrict__ bias, unsigned short* __restrict__ outSw,
    int OC, int mode, float scale) {
  const int HW = 4096, Cc = 512;
  __shared__ unsigned short tile[8][32][33];
  int w = threadIdx.x >> 5, lane = threadIdx.x & 31;
  int half = lane >> 4, l16 = lane & 15;
  int wave = (blockIdx.x << 3) + w;
  int oMac = OC >> 5;
  int perB = oMac * (HW >> 5);
  int b = wave / perB, rem = wave % perB;
  int n0 = (rem / oMac) << 5;
  int o0 = (rem % oMac) << 5;
  const float* Xb = X + (size_t)b * Cc * HW;

  v8f acc[2][2] = {};
  for (int kb = 0; kb < Cc / 32; ++kb) {
    int c0 = kb * 32;
    v16bf a[2], bb[2];
#pragma unroll
    for (int t = 0; t < 2; ++t)
#pragma unroll
      for (int i = 0; i < 16; ++i) {
        int ka = (i & 7) + ((i >> 3) << 4) + half * 8;  // A 16x32 layout
        a[t][i]  = (__bf16)W[(size_t)(o0 + t * 16 + l16) * Cc + c0 + ka];
        int kc = (i & 7) + ((i >> 3) << 4) + half * 8;  // B 32x16 layout
        bb[t][i] = (__bf16)Xb[(size_t)(c0 + kc) * HW + n0 + t * 16 + l16];
      }
#pragma unroll
    for (int ob = 0; ob < 2; ++ob)
#pragma unroll
      for (int nb = 0; nb < 2; ++nb)
        acc[ob][nb] = __builtin_amdgcn_wmma_f32_16x16x32_bf16(
            false, a[ob], false, bb[nb], (short)0, acc[ob][nb], false, false);
  }
  // bias + scale, f32 -> bf16, into per-wave LDS tile [o_loc][n_loc]
#pragma unroll
  for (int ob = 0; ob < 2; ++ob)
#pragma unroll
    for (int nb = 0; nb < 2; ++nb)
#pragma unroll
      for (int r = 0; r < 8; ++r) {
        int ol = ob * 16 + r + half * 8;
        tile[w][ol][nb * 16 + l16] =
            f32_to_bf16_bits((acc[ob][nb][r] + bias[o0 + ol]) * scale);
      }
  __syncthreads();

  unsigned short vv[16];
  if (mode == 0) {  // Q: A-fragment order (m = n-local, k = o-local)
#pragma unroll
    for (int nb = 0; nb < 2; ++nb) {
#pragma unroll
      for (int i = 0; i < 16; ++i) {
        int ko = (i & 7) + ((i >> 3) << 4) + half * 8;
        vv[i] = tile[w][ko][nb * 16 + l16];
      }
      size_t nt = (size_t)(n0 >> 4) + nb;
      store_frag(outSw + (((size_t)b * 256 + nt) * 2 + (o0 >> 5)) * 512 + lane * 16, vv);
    }
  } else if (mode == 1) {  // K: B-fragment order (k = o-local, n = m-local)
#pragma unroll
    for (int nb = 0; nb < 2; ++nb) {
#pragma unroll
      for (int i = 0; i < 16; ++i)
        vv[i] = tile[w][(i & 7) + ((i >> 3) << 4) + half * 8][nb * 16 + l16];
      size_t mt = (size_t)(n0 >> 4) + nb;
      store_frag(outSw + (((size_t)b * 256 + mt) * 2 + (o0 >> 5)) * 512 + lane * 16, vv);
    }
  } else {  // Vt: B-fragment order (k = m-local = n_loc, n = c-local = o_loc)
#pragma unroll
    for (int oc = 0; oc < 2; ++oc) {
#pragma unroll
      for (int i = 0; i < 16; ++i)
        vv[i] = tile[w][oc * 16 + l16][(i & 7) + ((i >> 3) << 4) + half * 8];
      size_t cn = (size_t)(o0 >> 4) + oc;
      store_frag(outSw + (((size_t)b * 128 + (n0 >> 5)) * 32 + cn) * 512 + lane * 16, vv);
    }
  }
}

// -------------------------------------------------------------------------
// Flash attention, all fragments loaded as contiguous 32B/lane vectors.
// Block = 16 query rows, 8 waves; 256-wide key tiles.
// -------------------------------------------------------------------------
#define MT 256

__global__ __launch_bounds__(256) void attn_kernel(
    const unsigned short* __restrict__ QSw, const unsigned short* __restrict__ KSw,
    const unsigned short* __restrict__ VSw, const float* __restrict__ value,
    const float* __restrict__ gamma, float* __restrict__ out) {
  const int HW = 4096, Cc = 512;
  __shared__ float sS[16][MT + 4];                       // S tile, padded
  __shared__ __align__(32) unsigned short sPfrag[8 * 512];  // P in A-frag order
  __shared__ float sFactor[16];

  int w = threadIdx.x >> 5, lane = threadIdx.x & 31;
  int half = lane >> 4, l16 = lane & 15;
  int b  = blockIdx.x >> 8;
  int nt = blockIdx.x & 255;
  int n0 = nt << 4;

  const unsigned short* Qb  = QSw + ((size_t)b * 256 + nt) * 2 * 512;
  const unsigned short* Kb0 = KSw + (size_t)b * 256 * 2 * 512;
  const unsigned short* Vb0 = VSw + (size_t)b * 128 * 32 * 512;

  v16bf aQ[2];
  aQ[0] = *(const v16bf*)(Qb + lane * 16);
  aQ[1] = *(const v16bf*)(Qb + 512 + lane * 16);

  v8f acc[4] = {};
  float rm[2] = {-INFINITY, -INFINITY};
  float rl[2] = {0.f, 0.f};

  for (int mt = 0; mt < HW / MT; ++mt) {
    int m0 = mt * MT;
    if (mt + 1 < HW / MT) {  // prefetch next 32KB K block (global_prefetch_b8)
      __builtin_prefetch(
          Kb0 + (size_t)((m0 + MT) >> 4) * 1024 + (size_t)threadIdx.x * 64, 0, 1);
    }
    // ---- phase 1: S[16,256] = Q K^T ----
#pragma unroll
    for (int s = 0; s < 2; ++s) {
      int mtile = (m0 >> 4) + w * 2 + s;
      v8f sacc = {};
#pragma unroll
      for (int kb = 0; kb < 2; ++kb) {
        v16bf bk = *(const v16bf*)(Kb0 + ((size_t)mtile * 2 + kb) * 512 + lane * 16);
        sacc = __builtin_amdgcn_wmma_f32_16x16x32_bf16(false, aQ[kb], false, bk,
                                                       (short)0, sacc, false, false);
      }
#pragma unroll
      for (int r = 0; r < 8; ++r) sS[r + half * 8][(w * 2 + s) * 16 + l16] = sacc[r];
    }
    __syncthreads();
    // ---- phase 2: online softmax (rows 2w, 2w+1); P -> A-frag order LDS ----
#pragma unroll
    for (int s = 0; s < 2; ++s) {
      int row = w * 2 + s;
      float vals[8];
      float tmax = -INFINITY;
#pragma unroll
      for (int c = 0; c < 8; ++c) {
        vals[c] = sS[row][c * 32 + lane];
        tmax = fmaxf(tmax, vals[c]);
      }
      for (int off = 16; off; off >>= 1) tmax = fmaxf(tmax, __shfl_xor(tmax, off, 32));
      float nm  = fmaxf(rm[s], tmax);
      float fac = __expf(rm[s] - nm);
      float tsum = 0.f;
      int half2 = (lane >> 3) & 1;                    // inverse A-layout mapping
      int i2    = (lane & 7) + ((lane >> 4) << 3);    // for k-in-32 = lane
#pragma unroll
      for (int c = 0; c < 8; ++c) {
        float p = __expf(vals[c] - nm);
        tsum += p;
        sPfrag[(c * 32 + half2 * 16 + row) * 16 + i2] = f32_to_bf16_bits(p);
      }
      for (int off = 16; off; off >>= 1) tsum += __shfl_xor(tsum, off, 32);
      rl[s] = rl[s] * fac + tsum;
      rm[s] = nm;
      if (lane == 0) sFactor[row] = fac;
    }
    __syncthreads();
    // ---- phase 3: rescale, O += P @ Vt (each wave owns 64 channels) ----
    float frow[8];
#pragma unroll
    for (int r = 0; r < 8; ++r) frow[r] = sFactor[r + half * 8];
#pragma unroll
    for (int cf = 0; cf < 4; ++cf)
#pragma unroll
      for (int r = 0; r < 8; ++r) acc[cf][r] *= frow[r];
    const unsigned short* vbase = Vb0 + (size_t)(m0 >> 5) * 32 * 512;
#pragma unroll 2
    for (int kb = 0; kb < 8; ++kb) {
      v16bf ap = *(const v16bf*)(sPfrag + kb * 512 + lane * 16);
#pragma unroll
      for (int cf = 0; cf < 4; ++cf) {
        v16bf bv = *(const v16bf*)(vbase + ((size_t)kb * 32 + w * 4 + cf) * 512 + lane * 16);
        acc[cf] = __builtin_amdgcn_wmma_f32_16x16x32_bf16(false, ap, false, bv,
                                                          (short)0, acc[cf], false, false);
      }
    }
    __syncthreads();
  }
  // ---- epilogue: normalize, gamma, residual; raw (HW,C)->(C,H,W) view ----
#pragma unroll
  for (int s = 0; s < 2; ++s)
    if (lane == 0) sFactor[w * 2 + s] = 1.0f / rl[s];
  __syncthreads();
  float inv[8];
#pragma unroll
  for (int r = 0; r < 8; ++r) inv[r] = sFactor[r + half * 8];
  float g = gamma[0];
  const float* valB = value + (size_t)b * Cc * HW;
  float*       outB = out   + (size_t)b * Cc * HW;
#pragma unroll
  for (int cf = 0; cf < 4; ++cf) {
    int cb = w * 64 + cf * 16;
#pragma unroll
    for (int r = 0; r < 8; ++r) {
      int n = n0 + r + half * 8;
      size_t f = (size_t)n * Cc + cb + l16;
      outB[f] = acc[cf][r] * inv[r] * g + valB[f];
    }
  }
}

extern "C" void kernel_launch(void* const* d_in, const int* in_sizes, int n_in,
                              void* d_out, int out_size, void* d_ws, size_t ws_size,
                              hipStream_t stream) {
  (void)in_sizes; (void)n_in; (void)out_size; (void)ws_size;
  const float* query = (const float*)d_in[0];
  const float* key   = (const float*)d_in[1];
  const float* value = (const float*)d_in[2];
  const float* Wq    = (const float*)d_in[3];
  const float* bq    = (const float*)d_in[4];
  const float* Wk    = (const float*)d_in[5];
  const float* bk    = (const float*)d_in[6];
  const float* Wv    = (const float*)d_in[7];
  const float* bv    = (const float*)d_in[8];
  const float* gamma = (const float*)d_in[9];
  float* out = (float*)d_out;

  // Workspace: QSw 4MB | KSw 4MB | VSw 32MB (all bf16, fragment-swizzled)
  unsigned short* QSw = (unsigned short*)d_ws;
  unsigned short* KSw = QSw + (size_t)8 * 256 * 2 * 512;
  unsigned short* VSw = KSw + (size_t)8 * 256 * 2 * 512;

  // Q/K: 8*2*128 = 2048 macro-tiles -> 256 blocks; V: 8*16*128 -> 2048 blocks
  proj_kernel<<<256,  256, 0, stream>>>(query, Wq, bq, QSw, 64,  0, 0.125f);
  proj_kernel<<<256,  256, 0, stream>>>(key,   Wk, bk, KSw, 64,  1, 1.0f);
  proj_kernel<<<2048, 256, 0, stream>>>(value, Wv, bv, VSw, 512, 2, 1.0f);
  attn_kernel<<<2048, 256, 0, stream>>>(QSw, KSw, VSw, value, gamma, out);
}
